// TransformerBlock_14499809592117
// MI455X (gfx1250) — matrix operand order repaired and checked by reading, not compiled, not run on hardware
//
#include <hip/hip_runtime.h>
#include <math.h>
#include <stdint.h>

typedef __bf16 bf16;
typedef __attribute__((ext_vector_type(16))) __bf16 v16bf;
typedef __attribute__((ext_vector_type(8)))  __bf16 v8bf;
typedef __attribute__((ext_vector_type(8)))  float  v8f;

#define BDIM 8
#define SEQ  1024
#define DM   512
#define NH   8
#define DKH  64
#define DFF_ 2048
#define TOK  (BDIM * SEQ)

union ABfrag { v16bf v; v8bf h[2]; };

__device__ __forceinline__ bf16 f2bf(float f) {
  unsigned u = __builtin_bit_cast(unsigned, f);
  u += 0x7FFFu + ((u >> 16) & 1u);
  unsigned short s = (unsigned short)(u >> 16);
  return __builtin_bit_cast(bf16, s);
}

__device__ __forceinline__ v8f zero8() {
  v8f z;
#pragma unroll
  for (int i = 0; i < 8; ++i) z[i] = 0.0f;
  return z;
}

__device__ __forceinline__ v8f bwmma(const ABfrag& a, const ABfrag& b, v8f c) {
  return __builtin_amdgcn_wmma_f32_16x16x32_bf16(false, a.v, false, b.v,
                                                 (short)0, c, false, false);
}

// Direct global -> LDS async copy (16B per lane), tracked by ASYNCcnt.
__device__ __forceinline__ void async_ld16(unsigned lds_off, const bf16* gaddr) {
  asm volatile("global_load_async_to_lds_b128 %0, %1, off"
               :: "v"(lds_off), "v"(gaddr) : "memory");
}
__device__ __forceinline__ void wait_async0() {
  asm volatile("s_wait_asynccnt 0x0" ::: "memory");
}
__device__ __forceinline__ void wait_async4() {
  asm volatile("s_wait_asynccnt 0x4" ::: "memory");
}

// ---------------- elementwise converts / transposes ----------------

__global__ __launch_bounds__(256)
void convert_bf16_kernel(const float* __restrict__ in, bf16* __restrict__ out, int n) {
  int i = blockIdx.x * 256 + threadIdx.x;
  if (i < n) out[i] = f2bf(in[i]);
}

// W [K,N] f32 -> Wt [N,K] bf16
__global__ __launch_bounds__(256)
void transpose_bf16_kernel(const float* __restrict__ W, bf16* __restrict__ Wt,
                           int K, int N) {
  int i = blockIdx.x * 256 + threadIdx.x;
  if (i >= K * N) return;
  int k = i / N, n = i % N;
  Wt[(size_t)n * K + k] = f2bf(W[i]);
}

// Vb [B*N, 512] bf16 -> Vtb [B, H, 64, N] bf16
__global__ __launch_bounds__(256)
void vtrans_kernel(const bf16* __restrict__ Vb, bf16* __restrict__ Vtb) {
  int i = blockIdx.x * 256 + threadIdx.x;
  int row = i >> 9;          // token
  int c = i & 511;
  int h = c >> 6, d = c & 63;
  int b = row >> 10, n = row & 1023;
  Vtb[(((size_t)(b * NH + h) * DKH + d) << 10) + n] = Vb[i];
}

// ---------------- WMMA GEMM: out = A[M,K] * Bt[N,K]^T + bias ----------------
// 64x64 block tile, 4 waves x (32x32), async double-buffered LDS staging.

__global__ __launch_bounds__(128)
void gemm_bf16_kernel(const bf16* __restrict__ A, const bf16* __restrict__ Bt,
                      const float* __restrict__ bias,
                      float* __restrict__ outF, bf16* __restrict__ outB,
                      int M, int N, int K, int act) {
  __shared__ __align__(16) bf16 As[2][64 * 40];
  __shared__ __align__(16) bf16 Bs[2][64 * 40];
  const int tid  = threadIdx.x;
  const int lane = tid & 31;
  const int wv   = tid >> 5;
  const int ln15 = lane & 15;
  const int hi   = (lane >> 4) & 1;
  const int br = blockIdx.y * 64;
  const int bc = blockIdx.x * 64;
  const int wr = (wv & 1) * 32;
  const int wc = (wv >> 1) * 32;
  const int srow = tid >> 1;
  const int scol = (tid & 1) << 4;

  const size_t arow  = (size_t)(br + srow) * K + scol;
  const size_t btrow = (size_t)(bc + srow) * K + scol;

  v8f c00 = zero8(), c01 = zero8(), c10 = zero8(), c11 = zero8();

  auto stage = [&](int k0, int sbuf) {
    const bf16* ag = A  + arow  + k0;
    const bf16* bg = Bt + btrow + k0;
    unsigned la = (unsigned)(uintptr_t)&As[sbuf][srow * 40 + scol];
    unsigned lb = (unsigned)(uintptr_t)&Bs[sbuf][srow * 40 + scol];
    async_ld16(la,      ag);
    async_ld16(la + 16, ag + 8);
    async_ld16(lb,      bg);
    async_ld16(lb + 16, bg + 8);
  };

  stage(0, 0);
  int buf = 0;
  for (int k0 = 0; k0 < K; k0 += 32) {
    if (k0 + 32 < K) { stage(k0 + 32, buf ^ 1); wait_async4(); }
    else             { wait_async0(); }
    __syncthreads();

    ABfrag a0, a1, b0, b1;
    const bf16* ap0 = &As[buf][(wr + ln15) * 40 + hi * 8];
    a0.h[0] = *(const v8bf*)ap0;        a0.h[1] = *(const v8bf*)(ap0 + 16);
    const bf16* ap1 = &As[buf][(wr + 16 + ln15) * 40 + hi * 8];
    a1.h[0] = *(const v8bf*)ap1;        a1.h[1] = *(const v8bf*)(ap1 + 16);
    const bf16* bp0 = &Bs[buf][(wc + ln15) * 40 + hi * 8];
    b0.h[0] = *(const v8bf*)bp0;        b0.h[1] = *(const v8bf*)(bp0 + 16);
    const bf16* bp1 = &Bs[buf][(wc + 16 + ln15) * 40 + hi * 8];
    b1.h[0] = *(const v8bf*)bp1;        b1.h[1] = *(const v8bf*)(bp1 + 16);

    c00 = bwmma(a0, b0, c00);
    c01 = bwmma(a0, b1, c01);
    c10 = bwmma(a1, b0, c10);
    c11 = bwmma(a1, b1, c11);
    __syncthreads();
    buf ^= 1;
  }

  v8f cs[2][2] = {{c00, c01}, {c10, c11}};
#pragma unroll
  for (int i = 0; i < 2; ++i)
#pragma unroll
    for (int j = 0; j < 2; ++j)
#pragma unroll
      for (int e = 0; e < 8; ++e) {
        int row = br + wr + i * 16 + e + hi * 8;
        int col = bc + wc + j * 16 + ln15;
        float v = cs[i][j][e] + bias[col];
        if (act == 1) v = 0.5f * v * (1.0f + erff(v * 0.70710678118654752f));
        size_t o = (size_t)row * N + col;
        if (outB) outB[o] = f2bf(v);
        else      outF[o] = v;
      }
}

// ---------------- flash attention: 1 wave per 16-query tile ----------------

__global__ __launch_bounds__(32)
void attn_kernel(const bf16* __restrict__ Qb, const bf16* __restrict__ Kb,
                 const bf16* __restrict__ Vt_, const float* __restrict__ bias_table,
                 bf16* __restrict__ ctx) {
  __shared__ __align__(16) bf16 Pls[16 * 32];
  const int lane = threadIdx.x;
  const int ln15 = lane & 15;
  const int hi   = lane >> 4;
  const int qt = blockIdx.x, h = blockIdx.y, b = blockIdx.z;
  const int q0 = qt * 16;

  const bf16* Qh = Qb + ((size_t)b * SEQ) * DM + h * DKH;
  const bf16* Kh = Kb + ((size_t)b * SEQ) * DM + h * DKH;
  const bf16* Vt = Vt_ + ((size_t)(b * NH + h) * DKH) * SEQ;

  ABfrag qa0, qa1;
  {
    const bf16* qr = Qh + (size_t)(q0 + ln15) * DM;
    qa0.h[0] = *(const v8bf*)(qr + hi * 8);
    qa0.h[1] = *(const v8bf*)(qr + 16 + hi * 8);
    qa1.h[0] = *(const v8bf*)(qr + 32 + hi * 8);
    qa1.h[1] = *(const v8bf*)(qr + 48 + hi * 8);
  }

  float mrow[8], lrow[8], corr[8];
  v8f acc0 = zero8(), acc1 = zero8(), acc2 = zero8(), acc3 = zero8();
#pragma unroll
  for (int e = 0; e < 8; ++e) { mrow[e] = -1e30f; lrow[e] = 0.0f; }

  for (int j = 0; j < SEQ / 32; ++j) {
    const int kvb = j * 32;
    v8f s0 = zero8(), s1 = zero8();
    {
      const bf16* kr0 = Kh + (size_t)(kvb + ln15) * DM + hi * 8;
      const bf16* kr1 = Kh + (size_t)(kvb + 16 + ln15) * DM + hi * 8;
      ABfrag kb0, kb1;
      kb0.h[0] = *(const v8bf*)(kr0);      kb0.h[1] = *(const v8bf*)(kr0 + 16);
      kb1.h[0] = *(const v8bf*)(kr1);      kb1.h[1] = *(const v8bf*)(kr1 + 16);
      s0 = bwmma(qa0, kb0, s0);
      s1 = bwmma(qa0, kb1, s1);
      kb0.h[0] = *(const v8bf*)(kr0 + 32); kb0.h[1] = *(const v8bf*)(kr0 + 48);
      kb1.h[0] = *(const v8bf*)(kr1 + 32); kb1.h[1] = *(const v8bf*)(kr1 + 48);
      s0 = bwmma(qa1, kb0, s0);
      s1 = bwmma(qa1, kb1, s1);
    }

#pragma unroll
    for (int e = 0; e < 8; ++e) {
      const int m = e + hi * 8;
      const int q = q0 + m;
      int r0 = kvb + ln15 - q;       r0 = min(255, max(-255, r0));
      int r1 = kvb + 16 + ln15 - q;  r1 = min(255, max(-255, r1));
      float t0 = s0[e] * 0.125f + bias_table[(r0 + 255) * NH + h];
      float t1 = s1[e] * 0.125f + bias_table[(r1 + 255) * NH + h];
      float mx = fmaxf(t0, t1);
      mx = fmaxf(mx, __shfl_xor(mx, 1, 32));
      mx = fmaxf(mx, __shfl_xor(mx, 2, 32));
      mx = fmaxf(mx, __shfl_xor(mx, 4, 32));
      mx = fmaxf(mx, __shfl_xor(mx, 8, 32));
      const float mn = fmaxf(mrow[e], mx);
      const float cr = __expf(mrow[e] - mn);
      t0 = __expf(t0 - mn);
      t1 = __expf(t1 - mn);
      float rs = t0 + t1;
      rs += __shfl_xor(rs, 1, 32);
      rs += __shfl_xor(rs, 2, 32);
      rs += __shfl_xor(rs, 4, 32);
      rs += __shfl_xor(rs, 8, 32);
      lrow[e] = lrow[e] * cr + rs;
      mrow[e] = mn;
      corr[e] = cr;
      Pls[m * 32 + ln15]      = f2bf(t0);
      Pls[m * 32 + 16 + ln15] = f2bf(t1);
    }
#pragma unroll
    for (int e = 0; e < 8; ++e) {
      acc0[e] *= corr[e]; acc1[e] *= corr[e];
      acc2[e] *= corr[e]; acc3[e] *= corr[e];
    }
    __syncthreads();

    ABfrag pa;
    const bf16* pr = &Pls[ln15 * 32 + hi * 8];
    pa.h[0] = *(const v8bf*)pr;
    pa.h[1] = *(const v8bf*)(pr + 16);

    ABfrag vf;
    const bf16* vr = Vt + (size_t)ln15 * SEQ + kvb + hi * 8;
    vf.h[0] = *(const v8bf*)vr;              vf.h[1] = *(const v8bf*)(vr + 16);
    acc0 = bwmma(pa, vf, acc0);
    vr += (size_t)16 * SEQ;
    vf.h[0] = *(const v8bf*)vr;              vf.h[1] = *(const v8bf*)(vr + 16);
    acc1 = bwmma(pa, vf, acc1);
    vr += (size_t)16 * SEQ;
    vf.h[0] = *(const v8bf*)vr;              vf.h[1] = *(const v8bf*)(vr + 16);
    acc2 = bwmma(pa, vf, acc2);
    vr += (size_t)16 * SEQ;
    vf.h[0] = *(const v8bf*)vr;              vf.h[1] = *(const v8bf*)(vr + 16);
    acc3 = bwmma(pa, vf, acc3);
    __syncthreads();
  }

#pragma unroll
  for (int e = 0; e < 8; ++e) {
    const int m = e + hi * 8;
    const float inv = 1.0f / lrow[e];
    size_t o = ((size_t)b * SEQ + q0 + m) * DM + h * DKH + ln15;
    ctx[o]      = f2bf(acc0[e] * inv);
    ctx[o + 16] = f2bf(acc1[e] * inv);
    ctx[o + 32] = f2bf(acc2[e] * inv);
    ctx[o + 48] = f2bf(acc3[e] * inv);
  }
}

// ---------------- residual + LayerNorm ----------------

__global__ __launch_bounds__(256)
void ln_kernel(const float* __restrict__ xin, const float* __restrict__ res,
               const float* __restrict__ g, const float* __restrict__ be,
               float* __restrict__ outF, bf16* __restrict__ outB) {
  const int row = blockIdx.x;
  const int tid = threadIdx.x;
  const int lane = tid & 31, wv = tid >> 5;
  const float* a = xin + (size_t)row * DM;
  const float* r = res + (size_t)row * DM;
  float v0 = a[tid] + r[tid];
  float v1 = a[tid + 256] + r[tid + 256];

  __shared__ float red[8];
  float s = v0 + v1;
  for (int m = 16; m >= 1; m >>= 1) s += __shfl_xor(s, m, 32);
  if (lane == 0) red[wv] = s;
  __syncthreads();
  float tot = 0.f;
  for (int i = 0; i < 8; ++i) tot += red[i];
  float mean = tot * (1.0f / DM);
  __syncthreads();

  float d0 = v0 - mean, d1 = v1 - mean;
  float vs = d0 * d0 + d1 * d1;
  for (int m = 16; m >= 1; m >>= 1) vs += __shfl_xor(vs, m, 32);
  if (lane == 0) red[wv] = vs;
  __syncthreads();
  float vtot = 0.f;
  for (int i = 0; i < 8; ++i) vtot += red[i];
  float rstd = rsqrtf(vtot * (1.0f / DM) + 1e-5f);

  float o0 = d0 * rstd * g[tid] + be[tid];
  float o1 = d1 * rstd * g[tid + 256] + be[tid + 256];
  outF[(size_t)row * DM + tid]       = o0;
  outF[(size_t)row * DM + tid + 256] = o1;
  if (outB) {
    outB[(size_t)row * DM + tid]       = f2bf(o0);
    outB[(size_t)row * DM + tid + 256] = f2bf(o1);
  }
}

// ---------------- host orchestration ----------------

extern "C" void kernel_launch(void* const* d_in, const int* in_sizes, int n_in,
                              void* d_out, int out_size, void* d_ws, size_t ws_size,
                              hipStream_t stream) {
  const float* x   = (const float*)d_in[0];
  const float* Wq  = (const float*)d_in[1];
  const float* bq  = (const float*)d_in[2];
  const float* Wk  = (const float*)d_in[3];
  const float* bk  = (const float*)d_in[4];
  const float* Wv  = (const float*)d_in[5];
  const float* bv  = (const float*)d_in[6];
  const float* Wo  = (const float*)d_in[7];
  const float* bo  = (const float*)d_in[8];
  const float* bt  = (const float*)d_in[9];
  const float* W1  = (const float*)d_in[10];
  const float* b1  = (const float*)d_in[11];
  const float* W2  = (const float*)d_in[12];
  const float* b2  = (const float*)d_in[13];
  const float* g1  = (const float*)d_in[14];
  const float* be1 = (const float*)d_in[15];
  const float* g2  = (const float*)d_in[16];
  const float* be2 = (const float*)d_in[17];

  char* ws = (char*)d_ws;
  size_t off = 0;
  auto alloc = [&](size_t bytes) -> char* {
    char* p = ws + off;
    off += (bytes + 255) & ~(size_t)255;
    return p;
  };

  bf16* xb   = (bf16*)alloc((size_t)TOK * DM * 2);
  bf16* Qb   = (bf16*)alloc((size_t)TOK * DM * 2);
  bf16* Kb   = (bf16*)alloc((size_t)TOK * DM * 2);
  bf16* Vb   = (bf16*)alloc((size_t)TOK * DM * 2);
  bf16* Vtb  = (bf16*)alloc((size_t)TOK * DM * 2);
  bf16* ctxb = (bf16*)alloc((size_t)TOK * DM * 2);
  bf16* WqT  = (bf16*)alloc((size_t)DM * DM * 2);
  bf16* WkT  = (bf16*)alloc((size_t)DM * DM * 2);
  bf16* WvT  = (bf16*)alloc((size_t)DM * DM * 2);
  bf16* WoT  = (bf16*)alloc((size_t)DM * DM * 2);
  bf16* W1T  = (bf16*)alloc((size_t)DM * DFF_ * 2);
  bf16* W2T  = (bf16*)alloc((size_t)DFF_ * DM * 2);
  float* tmp1 = (float*)alloc((size_t)TOK * DM * 4);
  float* x1   = (float*)alloc((size_t)TOK * DM * 4);
  bf16*  x1b  = (bf16*)alloc((size_t)TOK * DM * 2);
  bf16*  hb   = (bf16*)alloc((size_t)TOK * DFF_ * 2);
  float* tmp2 = (float*)alloc((size_t)TOK * DM * 4);

  convert_bf16_kernel<<<(TOK * DM) / 256, 256, 0, stream>>>(x, xb, TOK * DM);
  transpose_bf16_kernel<<<(DM * DM) / 256, 256, 0, stream>>>(Wq, WqT, DM, DM);
  transpose_bf16_kernel<<<(DM * DM) / 256, 256, 0, stream>>>(Wk, WkT, DM, DM);
  transpose_bf16_kernel<<<(DM * DM) / 256, 256, 0, stream>>>(Wv, WvT, DM, DM);
  transpose_bf16_kernel<<<(DM * DM) / 256, 256, 0, stream>>>(Wo, WoT, DM, DM);
  transpose_bf16_kernel<<<(DM * DFF_) / 256, 256, 0, stream>>>(W1, W1T, DM, DFF_);
  transpose_bf16_kernel<<<(DFF_ * DM) / 256, 256, 0, stream>>>(W2, W2T, DFF_, DM);

  dim3 gP(DM / 64, TOK / 64);
  gemm_bf16_kernel<<<gP, 128, 0, stream>>>(xb, WqT, bq, nullptr, Qb, TOK, DM, DM, 0);
  gemm_bf16_kernel<<<gP, 128, 0, stream>>>(xb, WkT, bk, nullptr, Kb, TOK, DM, DM, 0);
  gemm_bf16_kernel<<<gP, 128, 0, stream>>>(xb, WvT, bv, nullptr, Vb, TOK, DM, DM, 0);

  vtrans_kernel<<<(TOK * DM) / 256, 256, 0, stream>>>(Vb, Vtb);

  dim3 gA(SEQ / 16, NH, BDIM);
  attn_kernel<<<gA, 32, 0, stream>>>(Qb, Kb, Vtb, bt, ctxb);

  gemm_bf16_kernel<<<gP, 128, 0, stream>>>(ctxb, WoT, bo, tmp1, nullptr, TOK, DM, DM, 0);
  ln_kernel<<<TOK, 256, 0, stream>>>(x, tmp1, g1, be1, x1, x1b);

  dim3 gF1(DFF_ / 64, TOK / 64);
  gemm_bf16_kernel<<<gF1, 128, 0, stream>>>(x1b, W1T, b1, nullptr, hb, TOK, DFF_, DM, 1);
  gemm_bf16_kernel<<<gP, 128, 0, stream>>>(hb, W2T, b2, tmp2, nullptr, TOK, DM, DFF_, 0);
  ln_kernel<<<TOK, 256, 0, stream>>>(x1, tmp2, g2, be2, (float*)d_out, nullptr);
}
